// MultiHeadAttention_48077863912273
// MI455X (gfx1250) — matrix operand order repaired and checked by reading, not compiled, not run on hardware
//
#include <hip/hip_runtime.h>
#include <hip/hip_bf16.h>

#define D_MODEL 1024
#define NHEAD   16
#define DK      64
#define SEQ     2048
#define BATCH   2
#define NROWS   (BATCH * SEQ)   // 4096 total rows across batch

typedef __attribute__((ext_vector_type(16))) _Float16 v16h;
typedef __attribute__((ext_vector_type(8)))  _Float16 v8h;
typedef __attribute__((ext_vector_type(8)))  float    v8f;
typedef __attribute__((ext_vector_type(4)))  float    v4f;

// ---------------------------------------------------------------------------
// Fragment loaders per CDNA5 ISA 7.12.2 (wave32):
//  A 16x32 f16:  lanes 0-15 -> M=0..15, K 0..7 then 16..23; lanes 16-31 same M,
//                K 8..15 then 24..31.  Per lane: two contiguous 8-half runs.
//  B 32x16 f16:  lanes 0-15 -> N=lane, K 0..7 then 16..23; lanes 16-31 -> N=lane-16,
//                K 8..15 then 24..31.  Per lane: two 8-half runs of B^T row N.
//  C/D 16x16 f32: VGPR r -> M = (lane>=16 ? 8:0) + r, N = lane&15.
// ---------------------------------------------------------------------------

static __device__ inline v16h load_a_f16(const _Float16* __restrict__ tile, int lda, int lane) {
  const _Float16* p = tile + (size_t)(lane & 15) * lda + ((lane >> 4) * 8);
  v8h lo = *(const v8h*)(p);
  v8h hi = *(const v8h*)(p + 16);
  return __builtin_shufflevector(lo, hi, 0, 1, 2, 3, 4, 5, 6, 7, 8, 9, 10, 11, 12, 13, 14, 15);
}

// works for global or LDS-backed pointers (compiler infers address space)
static __device__ inline v16h load_a_f32(const float* tile, int lda, int lane) {
  const float* p = tile + (size_t)(lane & 15) * lda + ((lane >> 4) * 8);
  v4f x0 = *(const v4f*)(p);
  v4f x1 = *(const v4f*)(p + 4);
  v4f x2 = *(const v4f*)(p + 16);
  v4f x3 = *(const v4f*)(p + 20);
  v16h a;
#pragma unroll
  for (int i = 0; i < 4; ++i) {
    a[i]      = (_Float16)x0[i];
    a[4 + i]  = (_Float16)x1[i];
    a[8 + i]  = (_Float16)x2[i];
    a[12 + i] = (_Float16)x3[i];
  }
  return a;
}

// tile points at B^T[n0][k0]; ldb = row stride of B^T (the "n" major array)
static __device__ inline v16h load_bt_f16(const _Float16* __restrict__ tile, int ldb, int lane) {
  const _Float16* p = tile + (size_t)(lane & 15) * ldb + ((lane >> 4) * 8);
  v8h lo = *(const v8h*)(p);
  v8h hi = *(const v8h*)(p + 16);
  return __builtin_shufflevector(lo, hi, 0, 1, 2, 3, 4, 5, 6, 7, 8, 9, 10, 11, 12, 13, 14, 15);
}

static __device__ inline v8f wmma_f16(v16h a, v16h b, v8f c) {
  return __builtin_amdgcn_wmma_f32_16x16x32_f16(false, a, false, b, (short)0, c, false, false);
}

// ---------------------------------------------------------------------------
// Weight convert + transpose: W[k][n] fp32 -> Wt[n][k] f16
// ---------------------------------------------------------------------------
__global__ __launch_bounds__(256) void wcast_kernel(const float* __restrict__ W,
                                                    _Float16* __restrict__ Wt) {
  int idx = blockIdx.x * 256 + threadIdx.x;       // 0 .. 1024*1024-1
  int k = idx >> 10;
  int n = idx & (D_MODEL - 1);
  Wt[(size_t)n * D_MODEL + k] = (_Float16)W[idx];
}

// ---------------------------------------------------------------------------
// QKV projection: Y = X @ W + b, split into heads.
// vmode=0: Y[b][h][s][d] (Q,K);  vmode=1: Y[b][h][d][s] (V transposed)
// grid (16, 256), block (32,4): each wave -> one 16x16 tile, K-loop 32 WMMAs
// ---------------------------------------------------------------------------
__global__ __launch_bounds__(128) void proj_qkv_kernel(const float* __restrict__ X,
                                                       const _Float16* __restrict__ Wt,
                                                       const float* __restrict__ bias,
                                                       _Float16* __restrict__ Y, int vmode) {
  int lane = threadIdx.x;
  int n0 = (blockIdx.x * 4 + threadIdx.y) * 16;
  int m0 = blockIdx.y * 16;
  v8f acc = {};
  for (int k0 = 0; k0 < D_MODEL; k0 += 32) {
    v16h a = load_a_f32(X + (size_t)m0 * D_MODEL + k0, D_MODEL, lane);
    v16h b = load_bt_f16(Wt + (size_t)n0 * D_MODEL + k0, D_MODEL, lane);
    acc = wmma_f16(a, b, acc);
  }
  int n = n0 + (lane & 15);
  int h = n >> 6, d = n & (DK - 1);
  float bval = bias[n];
#pragma unroll
  for (int r = 0; r < 8; ++r) {
    int mg = m0 + ((lane >> 4) << 3) + r;         // global row 0..4095
    int bb = mg >> 11, s = mg & (SEQ - 1);
    float val = acc[r] + bval;
    size_t idx = vmode
        ? ((((size_t)bb * NHEAD + h) * DK + d) * SEQ + s)
        : ((((size_t)bb * NHEAD + h) * SEQ + s) * DK + d);
    Y[idx] = (_Float16)val;
  }
}

// ---------------------------------------------------------------------------
// Fused attention tail: scores -> softmax -> attn-out -> ctx, one q-tile of 16
// rows per workgroup.  Dynamic LDS: S[16][2048] fp32 scores (128 KB) -- two
// workgroups fit per 320 KB WGP.  attn hits HBM exactly once (output write).
// grid (1, 128, 32), block (32, 4).
// ---------------------------------------------------------------------------
__global__ __launch_bounds__(128) void attn_kernel(const _Float16* __restrict__ Qh,
                                                   const _Float16* __restrict__ Kh,
                                                   const _Float16* __restrict__ Vt,
                                                   float* __restrict__ attn,
                                                   _Float16* __restrict__ Ctx) {
  extern __shared__ __align__(16) char smem[];
  float* S = (float*)smem;                         // [16][SEQ]
  __shared__ float scratch[128];
  __shared__ float rowmax[16];
  __shared__ float rowinv[16];

  int lane = threadIdx.x;
  int wave = threadIdx.y;
  int tid  = wave * 32 + lane;
  int bh = blockIdx.z;
  int q0 = blockIdx.y * 16;

  const _Float16* Q = Qh + (size_t)bh * SEQ * DK;
  const _Float16* K = Kh + (size_t)bh * SEQ * DK;  // K[s][d] -> B^T rows
  const _Float16* V = Vt + (size_t)bh * DK * SEQ;  // V^T[d][s] -> B^T rows

  // ---- Phase 1: S = Q_tile @ K^T * 1/sqrt(dk) ---------------------------
  v16h a0 = load_a_f16(Q + (size_t)q0 * DK + 0,  DK, lane);
  v16h a1 = load_a_f16(Q + (size_t)q0 * DK + 32, DK, lane);
  for (int t = wave; t < SEQ / 16; t += 4) {       // key tiles
    int n0 = t * 16;
    v8f acc = {};
    acc = wmma_f16(a0, load_bt_f16(K + (size_t)n0 * DK + 0,  DK, lane), acc);
    acc = wmma_f16(a1, load_bt_f16(K + (size_t)n0 * DK + 32, DK, lane), acc);
    float* sp = S + (size_t)(((lane >> 4) << 3)) * SEQ + n0 + (lane & 15);
#pragma unroll
    for (int r = 0; r < 8; ++r) sp[(size_t)r * SEQ] = acc[r] * 0.125f;
  }
  __syncthreads();

  // ---- Phase 2: row softmax in LDS --------------------------------------
  int row = tid >> 3;                              // 16 rows x 8 threads
  int sub = tid & 7;
  float m = -3.4e38f;
  for (int i = sub; i < SEQ; i += 8) m = fmaxf(m, S[row * SEQ + i]);
  scratch[tid] = m;
  __syncthreads();
  if (tid < 16) {
    float v = scratch[tid * 8];
#pragma unroll
    for (int j = 1; j < 8; ++j) v = fmaxf(v, scratch[tid * 8 + j]);
    rowmax[tid] = v;
  }
  __syncthreads();
  m = rowmax[row];
  float sum = 0.f;
  for (int i = sub; i < SEQ; i += 8) {
    float e = __expf(S[row * SEQ + i] - m);
    S[row * SEQ + i] = e;
    sum += e;
  }
  scratch[tid] = sum;
  __syncthreads();
  if (tid < 16) {
    float v = scratch[tid * 8];
#pragma unroll
    for (int j = 1; j < 8; ++j) v += scratch[tid * 8 + j];
    rowinv[tid] = 1.f / v;
  }
  __syncthreads();

  // normalize in LDS + single coalesced write of attn to HBM
  float* gattn = attn + ((size_t)bh * SEQ + q0) * SEQ;
  for (int idx = tid; idx < 16 * SEQ; idx += 128) {
    int r = idx >> 11;                             // SEQ == 2048
    float val = S[idx] * rowinv[r];
    S[idx] = val;
    gattn[idx] = val;
  }
  __syncthreads();

  // ---- Phase 3: ctx = P @ V  (A fragments straight from LDS) ------------
  int n0 = wave * 16;                              // d tile within 64
  v8f acc = {};
  for (int k0 = 0; k0 < SEQ; k0 += 32) {
    v16h a = load_a_f32(S + k0, SEQ, lane);
    v16h b = load_bt_f16(V + (size_t)n0 * SEQ + k0, SEQ, lane);
    acc = wmma_f16(a, b, acc);
  }
  int bb = bh >> 4, h = bh & (NHEAD - 1);
  int n = n0 + (lane & 15);
#pragma unroll
  for (int r = 0; r < 8; ++r) {
    int s = q0 + ((lane >> 4) << 3) + r;
    Ctx[((size_t)bb * SEQ + s) * D_MODEL + h * DK + n] = (_Float16)acc[r];
  }
}

// ---------------------------------------------------------------------------
// out = Ctx @ Wo + bo  (fp32 output)
// grid (16, 256), block (32,4)
// ---------------------------------------------------------------------------
__global__ __launch_bounds__(128) void outproj_kernel(const _Float16* __restrict__ Ctx,
                                                      const _Float16* __restrict__ Wot,
                                                      const float* __restrict__ bo,
                                                      float* __restrict__ Out) {
  int lane = threadIdx.x;
  int n0 = (blockIdx.x * 4 + threadIdx.y) * 16;
  int m0 = blockIdx.y * 16;
  v8f acc = {};
  for (int k0 = 0; k0 < D_MODEL; k0 += 32) {
    v16h a = load_a_f16(Ctx + (size_t)m0 * D_MODEL + k0, D_MODEL, lane);
    v16h b = load_bt_f16(Wot + (size_t)n0 * D_MODEL + k0, D_MODEL, lane);
    acc = wmma_f16(a, b, acc);
  }
  int n = n0 + (lane & 15);
  float bval = bo[n];
#pragma unroll
  for (int r = 0; r < 8; ++r) {
    int mg = m0 + ((lane >> 4) << 3) + r;
    Out[(size_t)mg * D_MODEL + n] = acc[r] + bval;
  }
}

// ---------------------------------------------------------------------------
extern "C" void kernel_launch(void* const* d_in, const int* in_sizes, int n_in,
                              void* d_out, int out_size, void* d_ws, size_t ws_size,
                              hipStream_t stream) {
  (void)in_sizes; (void)n_in; (void)out_size; (void)ws_size;
  const float* q_in = (const float*)d_in[0];
  const float* k_in = (const float*)d_in[1];
  const float* v_in = (const float*)d_in[2];
  const float* Wq   = (const float*)d_in[3];
  const float* bq   = (const float*)d_in[4];
  const float* Wk   = (const float*)d_in[5];
  const float* bk   = (const float*)d_in[6];
  const float* Wv   = (const float*)d_in[7];
  const float* bv   = (const float*)d_in[8];
  const float* Wo   = (const float*)d_in[9];
  const float* bo   = (const float*)d_in[10];

  float* out  = (float*)d_out;                       // (B,S,1024)
  float* attn = out + (size_t)NROWS * D_MODEL;       // (B,H,S,S)

  char* ws = (char*)d_ws;
  const size_t HT = (size_t)NROWS * D_MODEL * sizeof(_Float16);   // 8 MiB
  const size_t HW = (size_t)D_MODEL * D_MODEL * sizeof(_Float16); // 2 MiB
  _Float16* Qh  = (_Float16*)(ws);
  _Float16* Kh  = (_Float16*)(ws + HT);
  _Float16* Vt  = (_Float16*)(ws + 2 * HT);
  _Float16* Ctx = (_Float16*)(ws + 3 * HT);
  _Float16* Wqt = (_Float16*)(ws + 4 * HT);
  _Float16* Wkt = (_Float16*)(ws + 4 * HT + HW);
  _Float16* Wvt = (_Float16*)(ws + 4 * HT + 2 * HW);
  _Float16* Wot = (_Float16*)(ws + 4 * HT + 3 * HW);

  int welems = D_MODEL * D_MODEL;
  wcast_kernel<<<welems / 256, 256, 0, stream>>>(Wq, Wqt);
  wcast_kernel<<<welems / 256, 256, 0, stream>>>(Wk, Wkt);
  wcast_kernel<<<welems / 256, 256, 0, stream>>>(Wv, Wvt);
  wcast_kernel<<<welems / 256, 256, 0, stream>>>(Wo, Wot);

  dim3 gblk(32, 4);
  dim3 pgrd(D_MODEL / 64, NROWS / 16);               // (16, 256)
  proj_qkv_kernel<<<pgrd, gblk, 0, stream>>>(q_in, Wqt, bq, Qh, 0);
  proj_qkv_kernel<<<pgrd, gblk, 0, stream>>>(k_in, Wkt, bk, Kh, 0);
  proj_qkv_kernel<<<pgrd, gblk, 0, stream>>>(v_in, Wvt, bv, Vt, 1);

  dim3 agrd(1, SEQ / 16, BATCH * NHEAD);             // (1, 128, 32)
  size_t smem_bytes = (size_t)16 * SEQ * sizeof(float);  // 128 KiB score tile
  attn_kernel<<<agrd, gblk, smem_bytes, stream>>>(Qh, Kh, Vt, attn, Ctx);

  dim3 ogrd(D_MODEL / 64, NROWS / 16);               // (16, 256)
  outproj_kernel<<<ogrd, gblk, 0, stream>>>(Ctx, Wot, bo, out);
}
